// Attention_40149354283630
// MI455X (gfx1250) — compile-verified, matrix operand order checked by reading, not run on hardware
//
#include <hip/hip_runtime.h>

typedef __attribute__((ext_vector_type(16))) __bf16 v16bf;
typedef __attribute__((ext_vector_type(8)))  __bf16 v8bf;
typedef __attribute__((ext_vector_type(4)))  __bf16 v4bf;
typedef __attribute__((ext_vector_type(8)))  float  v8f;
typedef __attribute__((ext_vector_type(4)))  unsigned int u32x4;
typedef __attribute__((ext_vector_type(8)))  int i32x8;
typedef __attribute__((ext_vector_type(4)))  int i32x4;

#define HID_C 1024
#define NH_C 16
#define HD_C 64
#define S_C 1024
#define B_C 8

__device__ __forceinline__ v16bf ldpair(const __bf16* p0, const __bf16* p1) {
  v8bf lo = *(const v8bf*)p0;
  v8bf hi = *(const v8bf*)p1;
  return __builtin_shufflevector(lo, hi, 0, 1, 2, 3, 4, 5, 6, 7,
                                 8, 9, 10, 11, 12, 13, 14, 15);
}

__device__ __forceinline__ v8f wmma_bf16(v16bf a, v16bf b, v8f c) {
  return __builtin_amdgcn_wmma_f32_16x16x32_bf16(false, a, false, b,
                                                 (short)0, c, false, false);
}

// ---------------------------------------------------------------------------
// fp32 -> bf16 elementwise convert (activations).
// ---------------------------------------------------------------------------
__global__ __launch_bounds__(256) void cvt_bf16_kernel(
    const float* __restrict__ src, __bf16* __restrict__ dst)
{
  size_t i = ((size_t)blockIdx.x * 256 + threadIdx.x) * 4;
  float4 v = *(const float4*)(src + i);
  v4bf b;
  b[0] = (__bf16)v.x; b[1] = (__bf16)v.y; b[2] = (__bf16)v.z; b[3] = (__bf16)v.w;
  *(v4bf*)(dst + i) = b;
}

// ---------------------------------------------------------------------------
// W (1024x1024 f32 row-major [k][n]) -> WT (bf16 column-major [n][k]).
// ---------------------------------------------------------------------------
__global__ __launch_bounds__(256) void transpose_w_kernel(
    const float* __restrict__ W, __bf16* __restrict__ WT)
{
  __shared__ float tile[32][33];
  const int n = blockIdx.x * 32 + threadIdx.x;
  const int k0 = blockIdx.y * 32;
#pragma unroll
  for (int i = 0; i < 4; ++i)
    tile[threadIdx.y + 8 * i][threadIdx.x] =
        W[(size_t)(k0 + threadIdx.y + 8 * i) * HID_C + n];
  __syncthreads();
  const int kk = k0 + threadIdx.x;
  const int nn = blockIdx.x * 32 + threadIdx.y;
#pragma unroll
  for (int i = 0; i < 4; ++i)
    WT[(size_t)(nn + 8 * i) * HID_C + kk] =
        (__bf16)tile[threadIdx.x][threadIdx.y + 8 * i];
}

// ---------------------------------------------------------------------------
// bf16 GEMM with 32x32 register blocking per wave. The 32x1024 bf16 A panel
// is DMA'd into LDS by the Tensor Data Mover (one TENSOR_LOAD_TO_LDS per
// block, iterate mode: 32 rows of 1024 elems, LDS row stride padded to 1040
// elems for conflict-free ds_load_b128 A-fragments).
// mode 0: bf16 head-major out [B,NH,S,HD] + 2D-RoPE (Q or K)
// mode 1: bf16 transposed   out [B,NH,HD,S]        (V)
// mode 2: fp32 row-major    out [B*S, HID]         (final projection)
// ---------------------------------------------------------------------------
#define XS_STRIDE 1040   // bf16 elems; row stride 2080B = 130*16B
__global__ __launch_bounds__(256) void gemm_bf16_kernel(
    const __bf16* __restrict__ A, const __bf16* __restrict__ WT,
    const float* __restrict__ bias, void* __restrict__ outv,
    const int* __restrict__ widthPtr, int mode)
{
  __shared__ __bf16 xs[32 * XS_STRIDE];          // 66,560 B (CDNA5: 320KB/WGP)
  const int wave = threadIdx.x >> 5;
  const int lane = threadIdx.x & 31;
  const int ln16 = lane & 15;
  const int hi   = lane >> 4;
  const int n0   = (blockIdx.x * 8 + wave) * 32; // wave's 32 output columns
  const int row0 = blockIdx.y * 32;              // block's 32 output rows

  if (wave == 0) {
    // ---- Tensor DMA descriptor (D#): 32 rows x 1024 bf16, iterate mode ----
    unsigned long long ga =
        (unsigned long long)(uintptr_t)(A + (size_t)row0 * HID_C);
    unsigned ldsOff = (unsigned)(uintptr_t)xs;   // LDS aperture: low 32 bits
    u32x4 g0;
    g0[0] = 1u;                                  // count=1 (valid descriptor)
    g0[1] = ldsOff;                              // lds_addr (bytes)
    g0[2] = (unsigned)(ga & 0xFFFFFFFFu);        // global_addr[31:0]
    g0[3] = (unsigned)((ga >> 32) & 0x1FFFFFFu) | (2u << 30); // [56:32]|type=2
    i32x8 g1;
    g1[0] = (1 << 16) | (1 << 19);               // data_size=2B, iterate_enable
    g1[1] = (int)(1024u << 16);                  // tensor_dim0 = 1024 (lo 16)
    g1[2] = (int)(8192u << 16);                  // dim0 hi=0 | tensor_dim1=8192
    g1[3] = (int)(1024u << 16);                  // dim1 hi=0 | tile_dim0=1024
    g1[4] = 1;                                   // tile_dim1=1, tile_dim2=0
    g1[5] = 1024;                                // tensor_dim0_stride = 1024
    g1[6] = 0;
    g1[7] = 0;
    i32x4 g2;
    g2[0] = 0;                                   // tensor_dim2
    g2[1] = XS_STRIDE;                           // lds_addr_increment / iter
    g2[2] = HID_C;                               // global_addr_increment / iter
    g2[3] = (int)(31u << 16);                    // iterate_count = 31 -> 32x
    i32x4 g3; g3[0] = 0; g3[1] = 0; g3[2] = 0; g3[3] = 0;
    i32x8 g4; g4[0] = 0; g4[1] = 0; g4[2] = 0; g4[3] = 0;
              g4[4] = 0; g4[5] = 0; g4[6] = 0; g4[7] = 0;
    __builtin_amdgcn_tensor_load_to_lds(g0, g1, g2, g3, g4, 0);
    __builtin_amdgcn_s_wait_tensorcnt(0);
  }
  __syncthreads();

  v8f acc[2][2] = {};
  const __bf16* a0 = xs + (size_t)ln16 * XS_STRIDE;        // M-tile 0
  const __bf16* a1 = a0 + 16 * XS_STRIDE;                  // M-tile 1
  const __bf16* w0 = WT + (size_t)(n0 + ln16) * HID_C;     // N-tile 0
  const __bf16* w1 = w0 + 16 * HID_C;                      // N-tile 1
#pragma unroll
  for (int t = 0; t < 32; ++t) {
    const int kb = 32 * t + 8 * hi;
    v16bf af0 = ldpair(a0 + kb, a0 + kb + 16);
    v16bf af1 = ldpair(a1 + kb, a1 + kb + 16);
    v16bf bf0 = ldpair(w0 + kb, w0 + kb + 16);
    v16bf bf1 = ldpair(w1 + kb, w1 + kb + 16);
    acc[0][0] = wmma_bf16(af0, bf0, acc[0][0]);
    acc[0][1] = wmma_bf16(af0, bf1, acc[0][1]);
    acc[1][0] = wmma_bf16(af1, bf0, acc[1][0]);
    acc[1][1] = wmma_bf16(af1, bf1, acc[1][1]);
  }

  const int w = (mode == 0) ? widthPtr[0] : 32;
#pragma unroll
  for (int nt = 0; nt < 2; ++nt) {
    const int col = n0 + nt * 16 + ln16;
    const float bcol = bias[col];
    int region = 0; float invf = 0.f;
    if (mode == 0) {
      int hc = col & (HD_C - 1);
      if (hc < 16)      { int p = hc >> 1;        region = 0; invf = __powf(10000.f, -(float)p * (2.f / 16.f)); }
      else if (hc < 40) { int p = (hc - 16) >> 1; region = 1; invf = __powf(10000.f, -(float)p * (2.f / 24.f)); }
      else              { int p = (hc - 40) >> 1; region = 2; invf = __powf(10000.f, -(float)p * (2.f / 24.f)); }
    }
    const float sgn = (col & 1) ? 1.f : -1.f;
#pragma unroll
    for (int mt = 0; mt < 2; ++mt) {
#pragma unroll
      for (int r = 0; r < 8; ++r) {
        float v = acc[mt][nt][r] + bcol;
        int rowm = row0 + mt * 16 + r + hi * 8;
        int s = rowm & (S_C - 1);
        if (mode == 0) {
          float posv = (region == 0) ? (float)s
                     : (region == 1) ? (float)(s % w)
                                     : (float)(s / w);
          float theta = posv * invf;
          float cth = __cosf(theta), sth = __sinf(theta);
          float partner = __shfl_xor(v, 1, 32);
          v = v * cth + partner * (sgn * sth);
        }
        int b = rowm >> 10;
        int h = col >> 6, hd = col & 63;
        if (mode == 2) {
          ((float*)outv)[(size_t)rowm * HID_C + col] = v;
        } else if (mode == 0) {   // head-major [bh][s][hd]
          ((__bf16*)outv)[(((size_t)(b * NH_C + h)) * S_C + s) * HD_C + hd] = (__bf16)v;
        } else {                  // mode 1: transposed [bh][hd][s]
          ((__bf16*)outv)[(((size_t)(b * NH_C + h)) * HD_C + hd) * S_C + s] = (__bf16)v;
        }
      }
    }
  }
}

// ---------------------------------------------------------------------------
// Flash attention, bf16 WMMA: one wave = 16 queries x HD=64, 32-key blocks.
// QK^T: 4 WMMAs/block, PV: 4 WMMAs/block. fp32 online softmax; P goes
// D-layout -> bf16 LDS -> A-layout.
// ---------------------------------------------------------------------------
#define P_STRIDE 40      // 32 + 8 bf16 pad; row stride 80B = 5*16B
__global__ __launch_bounds__(256) void attn_bf16_kernel(
    const __bf16* __restrict__ Q, const __bf16* __restrict__ K,
    const __bf16* __restrict__ VT, __bf16* __restrict__ AO)
{
  __shared__ __bf16 ldsP[8 * 16 * P_STRIDE];
  const int wave = threadIdx.x >> 5;
  const int lane = threadIdx.x & 31;
  const int ln16 = lane & 15;
  const int hi   = lane >> 4;
  const int bh   = blockIdx.y;
  const __bf16* Qb  = Q  + (size_t)bh * S_C * HD_C;   // [s][hd]
  const __bf16* Kb  = K  + (size_t)bh * S_C * HD_C;   // [s][hd]
  const __bf16* Vtb = VT + (size_t)bh * HD_C * S_C;   // [hd][s]
  const int q0 = (blockIdx.x * 8 + wave) * 16;
  __bf16* P = ldsP + wave * 16 * P_STRIDE;

  v16bf qa[2];
#pragma unroll
  for (int t = 0; t < 2; ++t) {
    const __bf16* qp = Qb + (size_t)(q0 + ln16) * HD_C + 32 * t + 8 * hi;
    qa[t] = ldpair(qp, qp + 16);
  }

  v8f o[4] = {};
  float m[8], l[8];
#pragma unroll
  for (int r = 0; r < 8; ++r) { m[r] = -3.0e38f; l[r] = 0.f; }

  for (int kt = 0; kt < 32; ++kt) {
    const int key0 = kt * 32;
    v8f sc0 = {}, sc1 = {};
#pragma unroll
    for (int t = 0; t < 2; ++t) {
      const __bf16* k0p = Kb + (size_t)(key0 + ln16) * HD_C + 32 * t + 8 * hi;
      const __bf16* k1p = k0p + 16 * HD_C;
      sc0 = wmma_bf16(qa[t], ldpair(k0p, k0p + 16), sc0);
      sc1 = wmma_bf16(qa[t], ldpair(k1p, k1p + 16), sc1);
    }
#pragma unroll
    for (int r = 0; r < 8; ++r) {
      float x0 = sc0[r] * 0.125f;            // 1/sqrt(64)
      float x1 = sc1[r] * 0.125f;
      float mx = fmaxf(x0, x1);
      mx = fmaxf(mx, __shfl_xor(mx, 1, 32));
      mx = fmaxf(mx, __shfl_xor(mx, 2, 32));
      mx = fmaxf(mx, __shfl_xor(mx, 4, 32));
      mx = fmaxf(mx, __shfl_xor(mx, 8, 32));
      float nm   = fmaxf(m[r], mx);
      float corr = __expf(m[r] - nm);
      float p0   = __expf(x0 - nm);
      float p1   = __expf(x1 - nm);
      float rs = p0 + p1;
      rs += __shfl_xor(rs, 1, 32);
      rs += __shfl_xor(rs, 2, 32);
      rs += __shfl_xor(rs, 4, 32);
      rs += __shfl_xor(rs, 8, 32);
      l[r] = l[r] * corr + rs;
      m[r] = nm;
#pragma unroll
      for (int nt = 0; nt < 4; ++nt) o[nt][r] *= corr;
      P[(r + hi * 8) * P_STRIDE + ln16]      = (__bf16)p0;
      P[(r + hi * 8) * P_STRIDE + 16 + ln16] = (__bf16)p1;
    }
    __builtin_amdgcn_wave_barrier();         // LDS store->load, same wave
    {
      const __bf16* pp = P + ln16 * P_STRIDE + 8 * hi;
      v16bf pa = ldpair(pp, pp + 16);
#pragma unroll
      for (int nt = 0; nt < 4; ++nt) {
        const __bf16* vp = Vtb + (size_t)(nt * 16 + ln16) * S_C + key0 + 8 * hi;
        o[nt] = wmma_bf16(pa, ldpair(vp, vp + 16), o[nt]);
      }
    }
    __builtin_amdgcn_wave_barrier();         // WAR before next P store
  }

  const int b = bh >> 4, h = bh & 15;
  float rl[8];
#pragma unroll
  for (int r = 0; r < 8; ++r) rl[r] = 1.f / l[r];
#pragma unroll
  for (int nt = 0; nt < 4; ++nt) {
#pragma unroll
    for (int r = 0; r < 8; ++r) {
      int q = q0 + r + hi * 8;
      int col = h * HD_C + nt * 16 + ln16;
      AO[((size_t)(b * S_C + q)) * HID_C + col] = (__bf16)(o[nt][r] * rl[r]);
    }
  }
}

// ---------------------------------------------------------------------------
extern "C" void kernel_launch(void* const* d_in, const int* in_sizes, int n_in,
                              void* d_out, int out_size, void* d_ws, size_t ws_size,
                              hipStream_t stream) {
  (void)in_sizes; (void)n_in; (void)out_size; (void)ws_size;
  const float* x  = (const float*)d_in[0];
  const float* Wq = (const float*)d_in[1];
  const float* bq = (const float*)d_in[2];
  const float* Wk = (const float*)d_in[3];
  const float* bk = (const float*)d_in[4];
  const float* Wv = (const float*)d_in[5];
  const float* bv = (const float*)d_in[6];
  const float* Wo = (const float*)d_in[7];
  const float* bo = (const float*)d_in[8];
  const int* width = (const int*)d_in[10];
  float* out = (float*)d_out;

  // bf16 workspace layout (~92 MB)
  const size_t NX = (size_t)B_C * S_C * HID_C;   // 8388608
  const size_t NW = (size_t)HID_C * HID_C;       // 1048576
  __bf16* xh  = (__bf16*)d_ws;
  __bf16* WqT = xh + NX;
  __bf16* WkT = WqT + NW;
  __bf16* WvT = WkT + NW;
  __bf16* WoT = WvT + NW;
  __bf16* Qh  = WoT + NW;          // [bh][s][hd]
  __bf16* Kh  = Qh + NX;           // [bh][s][hd]
  __bf16* Vt  = Kh + NX;           // [bh][hd][s]
  __bf16* AO  = Vt + NX;           // [b*s][hid]

  dim3 blk(256);
  cvt_bf16_kernel<<<dim3(NX / (256 * 4)), blk, 0, stream>>>(x, xh);
  transpose_w_kernel<<<dim3(32, 32), dim3(32, 8), 0, stream>>>(Wq, WqT);
  transpose_w_kernel<<<dim3(32, 32), dim3(32, 8), 0, stream>>>(Wk, WkT);
  transpose_w_kernel<<<dim3(32, 32), dim3(32, 8), 0, stream>>>(Wv, WvT);
  transpose_w_kernel<<<dim3(32, 32), dim3(32, 8), 0, stream>>>(Wo, WoT);

  dim3 gGemm(4, 256);              // 4 col-blocks-of-256 x 256 row-blocks-of-32
  gemm_bf16_kernel<<<gGemm, blk, 0, stream>>>(xh, WqT, bq, Qh, width, 0);
  gemm_bf16_kernel<<<gGemm, blk, 0, stream>>>(xh, WkT, bk, Kh, width, 0);
  gemm_bf16_kernel<<<gGemm, blk, 0, stream>>>(xh, WvT, bv, Vt, width, 1);
  attn_bf16_kernel<<<dim3(8, B_C * NH_C), blk, 0, stream>>>(Qh, Kh, Vt, AO);
  gemm_bf16_kernel<<<gGemm, blk, 0, stream>>>(AO, WoT, bo, out, width, 2);
}
